// VIT_38938173506177
// MI455X (gfx1250) — compile-verified, matrix operand order checked by reading, not run on hardware
//
#include <hip/hip_runtime.h>
#include <stdint.h>

// ---------------- model constants ----------------
#define DIM     768
#define NHEAD   12
#define DHEAD   64
#define NLAYER  12
#define NPATCH  196
#define NBATCH  64
#define NROWS   (NBATCH * NPATCH)   // 12544
#define DFF     (4 * DIM)           // 3072

typedef __attribute__((ext_vector_type(16))) __bf16        v16bf;
typedef __attribute__((ext_vector_type(8)))  float         v8f;
typedef __attribute__((ext_vector_type(2)))  unsigned int  u32x2;
typedef __attribute__((ext_vector_type(4)))  unsigned int  u32x4;
typedef __attribute__((ext_vector_type(8)))  int           i32x8;
typedef __attribute__((ext_vector_type(4)))  int           i32x4;

#if defined(__has_builtin)
#if __has_builtin(__builtin_amdgcn_tensor_load_to_lds)
#define HAVE_TDM 1
#endif
#endif
#ifndef HAVE_TDM
#define HAVE_TDM 0
#endif

__device__ __forceinline__ unsigned short f2bf(float f) {
  unsigned int u = __float_as_uint(f);
  u += 0x7FFFu + ((u >> 16) & 1u);           // round-to-nearest-even
  return (unsigned short)(u >> 16);
}

#if HAVE_TDM
// TDM: load a [rows=128, k=32] bf16 tile (row pitch = K elems in memory) into
// LDS with a 40-element (80B) padded row pitch via pad_enable(16dw interval,
// 4dw amount). D# per CDNA5 ISA ch.8.
__device__ __forceinline__ void tdm_load_tile(const unsigned short* gsrc,
                                              unsigned int ldsAddr,
                                              int K, int Nrows) {
  unsigned long long ga = (unsigned long long)(uintptr_t)gsrc;
  u32x4 g0;
  g0[0] = 1u;                                         // count=1 (valid D#)
  g0[1] = ldsAddr;                                    // lds_addr (bytes)
  g0[2] = (unsigned int)(ga & 0xFFFFFFFFu);           // global_addr lo
  g0[3] = (unsigned int)((ga >> 32) & 0x01FFFFFFu)    // global_addr hi (57b)
          | 0x80000000u;                              // type=2 ("image")
  i32x8 g1;
  g1[0] = (int)((1u << 16)      // data_size = 2 bytes
              | (1u << 20)      // pad_enable
              | (3u << 22)      // pad_interval = 16 DWORDs (64B = one k-row)
              | (3u << 25));    // pad_amount   = 4 DWORDs (16B -> 80B pitch)
  g1[1] = (int)(((unsigned)K & 0xFFFFu) << 16);                   // tensor_dim0 lo
  g1[2] = (int)(((unsigned)K >> 16) | (((unsigned)Nrows & 0xFFFFu) << 16));
  g1[3] = (int)((((unsigned)Nrows >> 16) & 0xFFFFu) | (32u << 16)); // tile_dim0=32
  g1[4] = (int)128u;                                  // tile_dim1=128, tile_dim2=0
  g1[5] = (int)K;                                     // tensor_dim0_stride lo32
  g1[6] = 0;
  g1[7] = 0;
  i32x4 z4; z4[0] = 0; z4[1] = 0; z4[2] = 0; z4[3] = 0;
#if __clang_major__ >= 23
  i32x8 z8; z8[0]=0; z8[1]=0; z8[2]=0; z8[3]=0; z8[4]=0; z8[5]=0; z8[6]=0; z8[7]=0;
  __builtin_amdgcn_tensor_load_to_lds(g0, g1, z4, z4, z8, 0);
#else
  __builtin_amdgcn_tensor_load_to_lds(g0, g1, z4, z4, 0);
#endif
}
#endif

// ---------------------------------------------------------------------------
// FAST GEMM (no bounds checks):  C = A x BT^T [+ resid]
//   A:  [M,K] bf16 row-major, BT: [N,K] bf16 row-major (pre-transposed B)
//   Requires M%128==0, N%128==0, K%32==0.
//   256 threads = 8 waves; block tile 128x128; wave tile 32x64 (8 WMMA/step).
//   B tile staged by the Tensor Data Mover (wave 0) when available.
// ---------------------------------------------------------------------------
__global__ __launch_bounds__(256)
void gemm_bf16_wmma_fast(const unsigned short* __restrict__ A,
                         const unsigned short* __restrict__ BT,
                         float* __restrict__ Cf,
                         unsigned short* __restrict__ Cb,
                         const float* __restrict__ Rz,
                         int M, int N, int K)
{
  constexpr int LP = 40;                     // 32 + 8 pad (80B pitch)
  __shared__ unsigned short As[128][LP];
  __shared__ unsigned short Bs[128][LP];

  const int row0 = blockIdx.y * 128;
  const int col0 = blockIdx.x * 128;
  const int tid  = threadIdx.x;
  const int lane = tid & 31;
  const int w    = tid >> 5;
  const int wm   = (w >> 1) * 32;
  const int wn   = (w & 1) * 64;

  const int sr  = tid >> 1;                  // staging row 0..127
  const int skc = (tid & 1) * 16;            // staging k offset 0/16
  const unsigned short* aSrc = A  + (long long)(row0 + sr) * K + skc;
  unsigned short* aDst = &As[sr][skc];
#if !HAVE_TDM
  const unsigned short* bSrc = BT + (long long)(col0 + sr) * K + skc;
  unsigned short* bDst = &Bs[sr][skc];
#endif

  v8f acc[2][4];
#pragma unroll
  for (int i = 0; i < 2; ++i)
#pragma unroll
    for (int j = 0; j < 4; ++j)
#pragma unroll
      for (int e = 0; e < 8; ++e) acc[i][j][e] = 0.0f;

  const int rrow  = lane & 15;
  const int khalf = (lane >> 4) * 8;

  for (int k0 = 0; k0 < K; k0 += 32) {
#if HAVE_TDM
    if (w == 0)
      tdm_load_tile(BT + (long long)col0 * K + k0,
                    (unsigned int)(uintptr_t)&Bs[0][0], K, N);
#endif
    {
      u32x4 a0 = *(const u32x4*)aSrc;
      u32x4 a1 = *(const u32x4*)(aSrc + 8);
      *(u32x4*)aDst = a0;
      *(u32x4*)(aDst + 8) = a1;
#if !HAVE_TDM
      u32x4 b0 = *(const u32x4*)bSrc;
      u32x4 b1 = *(const u32x4*)(bSrc + 8);
      *(u32x4*)bDst = b0;
      *(u32x4*)(bDst + 8) = b1;
      bSrc += 32;
#endif
    }
    aSrc += 32;
#if HAVE_TDM
    if (w == 0) __builtin_amdgcn_s_wait_tensorcnt(0);
#endif
    __syncthreads();

    union Frag { u32x4 u[2]; v16bf v; };
    Frag fb[4];
#pragma unroll
    for (int nt = 0; nt < 4; ++nt) {
      const unsigned short* bp = &Bs[wn + nt * 16 + rrow][0];
      fb[nt].u[0] = *(const u32x4*)(bp + khalf);
      fb[nt].u[1] = *(const u32x4*)(bp + 16 + khalf);
    }
#pragma unroll
    for (int mt = 0; mt < 2; ++mt) {
      Frag fa;
      const unsigned short* ap = &As[wm + mt * 16 + rrow][0];
      fa.u[0] = *(const u32x4*)(ap + khalf);
      fa.u[1] = *(const u32x4*)(ap + 16 + khalf);
#pragma unroll
      for (int nt = 0; nt < 4; ++nt)
        acc[mt][nt] = __builtin_amdgcn_wmma_f32_16x16x32_bf16(
            false, fa.v, false, fb[nt].v, (short)0, acc[mt][nt], false, false);
    }
    __syncthreads();
  }

  const int cn   = lane & 15;
  const int rsel = (lane >> 4) * 8;
#pragma unroll
  for (int mt = 0; mt < 2; ++mt)
#pragma unroll
    for (int nt = 0; nt < 4; ++nt) {
      const int gn = col0 + wn + nt * 16 + cn;
#pragma unroll
      for (int r = 0; r < 8; ++r) {
        const int gm = row0 + wm + mt * 16 + r + rsel;
        float v = acc[mt][nt][r];
        const long long ci = (long long)gm * N + gn;
        if (Rz) v += Rz[ci];
        if (Cf) Cf[ci] = v;
        if (Cb) Cb[ci] = f2bf(v);
      }
    }
}

// ---------------------------------------------------------------------------
// Generic bounds-checked WMMA GEMM (attention shapes).
// ---------------------------------------------------------------------------
template <bool TRANS_B>
__global__ __launch_bounds__(128)
void gemm_bf16_wmma(const unsigned short* __restrict__ A,
                    const unsigned short* __restrict__ B,
                    float* __restrict__ Cf,
                    unsigned short* __restrict__ Cb,
                    int M, int N, int K,
                    int lda, int ldb, int ldc,
                    int batchInner,
                    long long offAo, long long offAi,
                    long long offBo, long long offBi,
                    long long offCo, long long offCi,
                    float alpha)
{
  constexpr int BK = 32;
  constexpr int LP = BK + 8;
  __shared__ unsigned short As[64][LP];
  __shared__ unsigned short Bs[64][LP];

  const int z = blockIdx.z;
  const long long zo = z / batchInner, zi = z % batchInner;
  A += zo * offAo + zi * offAi;
  B += zo * offBo + zi * offBi;
  const long long coff = zo * offCo + zi * offCi;

  const int row0 = blockIdx.y * 64;
  const int col0 = blockIdx.x * 64;
  const int tid  = threadIdx.x;
  const int lane = tid & 31;
  const int w    = tid >> 5;
  const int wm   = (w >> 1) * 32;
  const int wn   = (w & 1) * 32;

  v8f acc[2][2];
#pragma unroll
  for (int i = 0; i < 2; ++i)
#pragma unroll
    for (int j = 0; j < 2; ++j)
#pragma unroll
      for (int e = 0; e < 8; ++e) acc[i][j][e] = 0.0f;

  for (int k0 = 0; k0 < K; k0 += BK) {
    {
      const int r  = tid >> 1;
      const int kc = (tid & 1) * 16;
      const int gr = row0 + r;
      unsigned short tmp[16];
#pragma unroll
      for (int i = 0; i < 16; ++i) tmp[i] = 0;
      if (gr < M) {
        const unsigned short* src = A + (long long)gr * lda + k0 + kc;
        if (k0 + kc + 16 <= K) {
          const u32x2* s = (const u32x2*)src;
          *(u32x2*)&tmp[0]  = s[0];
          *(u32x2*)&tmp[4]  = s[1];
          *(u32x2*)&tmp[8]  = s[2];
          *(u32x2*)&tmp[12] = s[3];
        } else {
          for (int i = 0; i < 16; ++i)
            if (k0 + kc + i < K) tmp[i] = src[i];
        }
      }
#pragma unroll
      for (int i = 0; i < 16; ++i) As[r][kc + i] = tmp[i];
    }
    if (TRANS_B) {
      const int n  = tid >> 1;
      const int kc = (tid & 1) * 16;
      const int gn = col0 + n;
      unsigned short tmp[16];
#pragma unroll
      for (int i = 0; i < 16; ++i) tmp[i] = 0;
      if (gn < N) {
        const unsigned short* src = B + (long long)gn * ldb + k0 + kc;
        if (k0 + kc + 16 <= K) {
          const u32x2* s = (const u32x2*)src;
          *(u32x2*)&tmp[0]  = s[0];
          *(u32x2*)&tmp[4]  = s[1];
          *(u32x2*)&tmp[8]  = s[2];
          *(u32x2*)&tmp[12] = s[3];
        } else {
          for (int i = 0; i < 16; ++i)
            if (k0 + kc + i < K) tmp[i] = src[i];
        }
      }
#pragma unroll
      for (int i = 0; i < 16; ++i) Bs[n][kc + i] = tmp[i];
    } else {
      const int kk = tid & 31;
      const int nb = (tid >> 5) * 16;
      const int gk = k0 + kk;
      unsigned short tmp[16];
#pragma unroll
      for (int i = 0; i < 16; ++i) tmp[i] = 0;
      if (gk < K) {
        const unsigned short* src = B + (long long)gk * ldb + col0 + nb;
        if (col0 + nb + 16 <= N) {
          const u32x2* s = (const u32x2*)src;
          *(u32x2*)&tmp[0]  = s[0];
          *(u32x2*)&tmp[4]  = s[1];
          *(u32x2*)&tmp[8]  = s[2];
          *(u32x2*)&tmp[12] = s[3];
        } else {
          for (int i = 0; i < 16; ++i)
            if (col0 + nb + i < N) tmp[i] = src[i];
        }
      }
#pragma unroll
      for (int i = 0; i < 16; ++i) Bs[nb + i][kk] = tmp[i];
    }
    __syncthreads();

    if (k0 + BK < K)
      __builtin_prefetch(A + (long long)(row0 + (tid >> 1)) * lda + (k0 + BK), 0, 1);

    const int rrow  = lane & 15;
    const int khalf = (lane >> 4) * 8;

    union Frag { u32x4 u[2]; v16bf v; };
    Frag fb[2];
#pragma unroll
    for (int nt = 0; nt < 2; ++nt) {
      const unsigned short* bp = &Bs[wn + nt * 16 + rrow][0];
      fb[nt].u[0] = *(const u32x4*)(bp + khalf);
      fb[nt].u[1] = *(const u32x4*)(bp + 16 + khalf);
    }
#pragma unroll
    for (int mt = 0; mt < 2; ++mt) {
      Frag fa;
      const unsigned short* ap = &As[wm + mt * 16 + rrow][0];
      fa.u[0] = *(const u32x4*)(ap + khalf);
      fa.u[1] = *(const u32x4*)(ap + 16 + khalf);
#pragma unroll
      for (int nt = 0; nt < 2; ++nt) {
        acc[mt][nt] = __builtin_amdgcn_wmma_f32_16x16x32_bf16(
            false, fa.v, false, fb[nt].v, (short)0, acc[mt][nt], false, false);
      }
    }
    __syncthreads();
  }

  const int cn   = lane & 15;
  const int rsel = (lane >> 4) * 8;
#pragma unroll
  for (int mt = 0; mt < 2; ++mt)
#pragma unroll
    for (int nt = 0; nt < 2; ++nt) {
      const int gn = col0 + wn + nt * 16 + cn;
      if (gn >= N) continue;
#pragma unroll
      for (int r = 0; r < 8; ++r) {
        const int gm = row0 + wm + mt * 16 + r + rsel;
        if (gm >= M) continue;
        float v = acc[mt][nt][r] * alpha;
        const long long ci = coff + (long long)gm * ldc + gn;
        if (Cf) Cf[ci] = v;
        if (Cb) Cb[ci] = f2bf(v);
      }
    }
}

// ---------------------------------------------------------------------------
// Elementwise / reduction helper kernels
// ---------------------------------------------------------------------------

__global__ __launch_bounds__(256)
void patchify_kernel(const float* __restrict__ x, unsigned short* __restrict__ xp)
{
  long long idx = (long long)blockIdx.x * 256 + threadIdx.x;
  if (idx >= (long long)NROWS * DIM) return;
  int k = (int)(idx % DIM);
  long long rn = idx / DIM;
  int n = (int)(rn % NPATCH);
  int b = (int)(rn / NPATCH);
  int ph = n / 14, pw = n % 14;
  int py = k / 48, rem = k % 48;
  int px = rem / 3, c = rem % 3;
  long long src = (((long long)b * 3 + c) * 224 + (ph * 16 + py)) * 224 + (pw * 16 + px);
  xp[idx] = f2bf(x[src]);
}

// in [K,N] fp32 -> out [N,K] bf16 (transposed cast)
__global__ __launch_bounds__(256)
void cast_t_kernel(const float* __restrict__ in, unsigned short* __restrict__ out,
                   int K, int N)
{
  long long i = (long long)blockIdx.x * 256 + threadIdx.x;
  if (i >= (long long)K * N) return;
  int n = (int)(i / K), k = (int)(i % K);
  out[i] = f2bf(in[(long long)k * N + n]);
}

// per-head weight [H, D, DH] fp32 -> transposed full [H*DH, D] bf16
__global__ __launch_bounds__(256)
void conv_headw_t_kernel(const float* __restrict__ in, unsigned short* __restrict__ out)
{
  int i = blockIdx.x * 256 + threadIdx.x;
  if (i >= DIM * DIM) return;
  int c = i / DIM, d = i % DIM;    // out[c][d], c = h*64+e
  out[i] = f2bf(in[((long long)(c >> 6) * DIM + d) * DHEAD + (c & 63)]);
}

__global__ __launch_bounds__(256)
void addpos_kernel(float* __restrict__ h, const float* __restrict__ pos,
                   unsigned short* __restrict__ hb)
{
  long long i = (long long)blockIdx.x * 256 + threadIdx.x;
  if (i >= (long long)NROWS * DIM) return;
  int d = (int)(i % DIM);
  int n = (int)((i / DIM) % NPATCH);
  float v = h[i] + pos[(long long)n * DIM + d];
  h[i] = v;
  hb[i] = f2bf(v);
}

__global__ __launch_bounds__(256)
void softmax_kernel(const float* __restrict__ s, unsigned short* __restrict__ p, int ncols)
{
  long long row = blockIdx.x;
  const float* src = s + row * ncols;
  unsigned short* dst = p + row * ncols;
  int tid = threadIdx.x;
  float v = (tid < ncols) ? src[tid] : -3.4e38f;
  __shared__ float red[256];
  red[tid] = v; __syncthreads();
  for (int st = 128; st > 0; st >>= 1) {
    if (tid < st) red[tid] = fmaxf(red[tid], red[tid + st]);
    __syncthreads();
  }
  float mx = red[0]; __syncthreads();
  float e = (tid < ncols) ? __expf(v - mx) : 0.0f;
  red[tid] = e; __syncthreads();
  for (int st = 128; st > 0; st >>= 1) {
    if (tid < st) red[tid] += red[tid + st];
    __syncthreads();
  }
  float inv = 1.0f / red[0];
  if (tid < ncols) dst[tid] = f2bf(e * inv);
}

__global__ __launch_bounds__(256)
void add_layernorm_kernel(const float* __restrict__ a, const float* __restrict__ b,
                          float* __restrict__ of, unsigned short* __restrict__ ob)
{
  long long row = blockIdx.x;
  const float* pa = a + row * DIM;
  const float* pb = b + row * DIM;
  int tid = threadIdx.x;
  float v0 = pa[tid]       + pb[tid];
  float v1 = pa[tid + 256] + pb[tid + 256];
  float v2 = pa[tid + 512] + pb[tid + 512];
  float s  = v0 + v1 + v2;
  float s2 = v0 * v0 + v1 * v1 + v2 * v2;
  __shared__ float r1[256], r2[256];
  r1[tid] = s; r2[tid] = s2; __syncthreads();
  for (int st = 128; st > 0; st >>= 1) {
    if (tid < st) { r1[tid] += r1[tid + st]; r2[tid] += r2[tid + st]; }
    __syncthreads();
  }
  float mu  = r1[0] * (1.0f / DIM);
  float var = r2[0] * (1.0f / DIM) - mu * mu;
  float inv = rsqrtf(var + 1e-5f);
  float* od = of + row * DIM;
  unsigned short* obd = ob + row * DIM;
  float o0 = (v0 - mu) * inv, o1 = (v1 - mu) * inv, o2 = (v2 - mu) * inv;
  od[tid] = o0;        obd[tid] = f2bf(o0);
  od[tid + 256] = o1;  obd[tid + 256] = f2bf(o1);
  od[tid + 512] = o2;  obd[tid + 512] = f2bf(o2);
}

// ---------------------------------------------------------------------------
// Launch: full 12-layer ViT forward
// ---------------------------------------------------------------------------
extern "C" void kernel_launch(void* const* d_in, const int* in_sizes, int n_in,
                              void* d_out, int out_size, void* d_ws, size_t ws_size,
                              hipStream_t stream)
{
  (void)in_sizes; (void)n_in; (void)out_size; (void)ws_size;
  const float* x    = (const float*)d_in[0];
  const float* proj = (const float*)d_in[1];
  const float* pos  = (const float*)d_in[2];
  const float* qw   = (const float*)d_in[3];
  const float* kw   = (const float*)d_in[4];
  const float* vw   = (const float*)d_in[5];
  const float* ow   = (const float*)d_in[6];
  const float* f1   = (const float*)d_in[7];
  const float* f2   = (const float*)d_in[8];
  float* outp = (float*)d_out;

  char* base = (char*)d_ws;
  size_t off = 0;
  auto take = [&](size_t bytes) -> char* {
    char* p = base + off;
    off += (bytes + 255) & ~(size_t)255;
    return p;
  };
  const size_t actF = (size_t)NROWS * DIM * 4;
  const size_t actB = (size_t)NROWS * DIM * 2;

  float*          hbuf    = (float*)take(actF);
  unsigned short* hbf     = (unsigned short*)take(actB);   // also patch matrix
  unsigned short* qbf     = (unsigned short*)take(actB);
  unsigned short* kbf     = (unsigned short*)take(actB);
  unsigned short* vbf     = (unsigned short*)take(actB);
  unsigned short* ctxbf   = (unsigned short*)take(actB);
  float*          inter   = (float*)take(actF);
  unsigned short* interbf = (unsigned short*)take(actB);
  unsigned short* wqbT    = (unsigned short*)take((size_t)DIM * DIM * 2);
  unsigned short* wkbT    = (unsigned short*)take((size_t)DIM * DIM * 2);
  unsigned short* wvbT    = (unsigned short*)take((size_t)DIM * DIM * 2);
  unsigned short* wobT    = (unsigned short*)take((size_t)DIM * DIM * 2);
  unsigned short* projT   = (unsigned short*)take((size_t)DIM * DIM * 2);
  unsigned short* f1T     = (unsigned short*)take((size_t)DIM * DFF * 2);
  unsigned short* f2T     = (unsigned short*)take((size_t)DIM * DFF * 2);

  const size_t scoresBytes = (size_t)NBATCH * NHEAD * NPATCH * NPATCH * 4;
  const size_t tBytes      = (size_t)NROWS * DFF * 2;
  char* big = take(scoresBytes > tBytes ? scoresBytes : tBytes);
  float*          scores = (float*)big;
  float*          oout   = (float*)big;           // live after scores are dead
  unsigned short* tbf    = (unsigned short*)big;  // live after oout is dead
  unsigned short* probs  = (unsigned short*)take(scoresBytes / 2);

  auto gemmFast = [&](const unsigned short* A, const unsigned short* BT,
                      float* Cf, unsigned short* Cb, const float* Rz,
                      int M, int N, int K) {
    dim3 g(N / 128, M / 128, 1);
    gemm_bf16_wmma_fast<<<g, dim3(256), 0, stream>>>(A, BT, Cf, Cb, Rz, M, N, K);
  };
  auto gemmNT = [&](const unsigned short* A, const unsigned short* B,
                    float* Cf, unsigned short* Cb,
                    int M, int N, int K, int lda, int ldb, int ldc,
                    int nz, int bi,
                    long long ao, long long ai, long long bo, long long bi2,
                    long long co, long long ci, float alpha) {
    dim3 g((N + 63) / 64, (M + 63) / 64, nz);
    gemm_bf16_wmma<true><<<g, dim3(128), 0, stream>>>(
        A, B, Cf, Cb, M, N, K, lda, ldb, ldc, bi, ao, ai, bo, bi2, co, ci, alpha);
  };
  auto gemmNN = [&](const unsigned short* A, const unsigned short* B,
                    float* Cf, unsigned short* Cb,
                    int M, int N, int K, int lda, int ldb, int ldc,
                    int nz, int bi,
                    long long ao, long long ai, long long bo, long long bi2,
                    long long co, long long ci, float alpha) {
    dim3 g((N + 63) / 64, (M + 63) / 64, nz);
    gemm_bf16_wmma<false><<<g, dim3(128), 0, stream>>>(
        A, B, Cf, Cb, M, N, K, lda, ldb, ldc, bi, ao, ai, bo, bi2, co, ci, alpha);
  };

  const long long nPE = (long long)NROWS * DIM;
  const int gDD  = (DIM * DIM + 255) / 256;
  const int gDF  = (int)(((long long)DIM * DFF + 255) / 256);

  // ---- patch embedding ----
  patchify_kernel<<<(unsigned)((nPE + 255) / 256), 256, 0, stream>>>(x, hbf);
  cast_t_kernel<<<gDD, 256, 0, stream>>>(proj, projT, DIM, DIM);
  gemmFast(hbf, projT, hbuf, nullptr, nullptr, NROWS, DIM, DIM);
  addpos_kernel<<<(unsigned)((nPE + 255) / 256), 256, 0, stream>>>(hbuf, pos, hbf);

  const long long wqkvStride = (long long)NHEAD * DIM * DHEAD;
  const long long sPerB = (long long)NHEAD * NPATCH * NPATCH;
  const long long sPerH = (long long)NPATCH * NPATCH;
  const long long aPerB = (long long)NPATCH * DIM;

  for (int l = 0; l < NLAYER; ++l) {
    // per-layer weight conversion: transposed bf16
    conv_headw_t_kernel<<<gDD, 256, 0, stream>>>(qw + l * wqkvStride, wqbT);
    conv_headw_t_kernel<<<gDD, 256, 0, stream>>>(kw + l * wqkvStride, wkbT);
    conv_headw_t_kernel<<<gDD, 256, 0, stream>>>(vw + l * wqkvStride, wvbT);
    cast_t_kernel<<<gDD, 256, 0, stream>>>(ow + (long long)l * DIM * DIM, wobT, DIM, DIM);
    cast_t_kernel<<<gDF, 256, 0, stream>>>(f1 + (long long)l * DIM * DFF, f1T, DIM, DFF);
    cast_t_kernel<<<gDF, 256, 0, stream>>>(f2 + (long long)l * DIM * DFF, f2T, DFF, DIM);

    // Q, K, V projections (bf16 outputs, [B,N,D] head-blocked columns)
    gemmFast(hbf, wqbT, nullptr, qbf, nullptr, NROWS, DIM, DIM);
    gemmFast(hbf, wkbT, nullptr, kbf, nullptr, NROWS, DIM, DIM);
    gemmFast(hbf, wvbT, nullptr, vbf, nullptr, NROWS, DIM, DIM);

    // scores = (Q K^T) * 1/sqrt(DH), batched over (b,h)
    gemmNT(qbf, kbf, scores, nullptr,
           NPATCH, NPATCH, DHEAD, DIM, DIM, NPATCH,
           NBATCH * NHEAD, NHEAD,
           aPerB, DHEAD, aPerB, DHEAD, sPerB, sPerH, 0.125f);

    softmax_kernel<<<NBATCH * NHEAD * NPATCH, 256, 0, stream>>>(scores, probs, NPATCH);

    // ctx = P V, back into [B,N,D] head-blocked layout (bf16)
    gemmNN(probs, vbf, nullptr, ctxbf,
           NPATCH, DHEAD, NPATCH, NPATCH, DIM, DIM,
           NBATCH * NHEAD, NHEAD,
           sPerB, sPerH, aPerB, DHEAD, aPerB, DHEAD, 1.0f);

    // out = ctx @ Wo
    gemmFast(ctxbf, wobT, oout, nullptr, nullptr, NROWS, DIM, DIM);

    // inter = LN(out + h)
    add_layernorm_kernel<<<NROWS, 256, 0, stream>>>(oout, hbuf, inter, interbf);

    // FFN (no activation): t = inter @ f1 ; h' = t @ f2 + inter
    gemmFast(interbf, f1T, nullptr, tbf, nullptr, NROWS, DFF, DIM);

    float* hdst = (l == NLAYER - 1) ? outp : hbuf;
    gemmFast(tbf, f2T, hdst, hbf, inter, NROWS, DIM, DFF);
  }
}
